// AttnBlock_86655260164414
// MI455X (gfx1250) — compile-verified, matrix operand order checked by reading, not run on hardware
//
#include <hip/hip_runtime.h>
#include <hip/hip_bf16.h>

#define DEVINL __device__ __forceinline__

typedef __attribute__((ext_vector_type(16))) __bf16       v16bf;
typedef __attribute__((ext_vector_type(8)))  float        v8f;
typedef __attribute__((ext_vector_type(8)))  unsigned int v8u;

static constexpr int Bn   = 16;
static constexpr int Cc   = 256;
static constexpr int Ll   = 2048;
static constexpr int ZDim = 128;
static constexpr int TE   = 512;
static constexpr int LP   = Ll + 16;  // padded stride for conv inputs
static constexpr int POFF = 8;        // halo offset inside padded row

// ---------------- helpers ----------------

DEVINL unsigned short f2bf(float f) {
  unsigned int u = __builtin_bit_cast(unsigned int, f);
  u += 0x7FFFu + ((u >> 16) & 1u);   // round-to-nearest-even
  return (unsigned short)(u >> 16);
}

DEVINL float silu_f(float v) { return v / (1.0f + __expf(-v)); }

// LDS byte offset of a __shared__ object (flat LDS aperture: addr[31:0]).
DEVINL unsigned lds_off32(const void* p) {
  return (unsigned)(unsigned long long)(uintptr_t)p;
}

// Load a 16-bit WMMA A/B fragment for one lane.
// row points at 16-bit elements of one matrix row (A: row m, B^T: col n);
// K mapping per CDNA5 ISA: VGPR r holds K = (r>=4)*16 + g*8 + (r&3)*2, +1.
// k0 even and grouped -> compiler fuses into two b128 loads.
DEVINL v16bf load_frag(const unsigned short* row, int g) {
  v8u u;
#pragma unroll
  for (int r = 0; r < 8; ++r) {
    int k0 = ((r & 4) << 2) + (g << 3) + ((r & 3) << 1);
    u[r] = *(const unsigned int*)(row + k0);
  }
  return __builtin_bit_cast(v16bf, u);
}

DEVINL v8f wmma_bf16(v16bf a, v16bf b, v8f c) {
  // D = A(16x32 bf16) * B(32x16 bf16) + C(16x16 f32)
  return __builtin_amdgcn_wmma_f32_16x16x32_bf16(false, a, false, b,
                                                 (short)0, c, false, false);
}

// ---------------- timestep embedding + projections ----------------
__global__ __launch_bounds__(256) void temb_kernel(
    const int* __restrict__ t, const float* __restrict__ z0,
    const float* __restrict__ zt, const float* __restrict__ tw1,
    const float* __restrict__ tb1, const float* __restrict__ tw2,
    const float* __restrict__ tb2, const float* __restrict__ tpw,
    const float* __restrict__ tpb, const float* __restrict__ zpw,
    const float* __restrict__ zpb, float* __restrict__ addout) {
  __shared__ float emb[128];
  __shared__ float t1[TE];
  __shared__ float t2s[TE];
  __shared__ float z0s[ZDim];
  __shared__ float zts[ZDim];
  const int b = blockIdx.x, tid = threadIdx.x;

  if (tid < 64) {
    float freq = __expf((float)tid * (-logf(10000.0f) / 63.0f));
    float arg = (float)t[b] * freq;
    emb[tid] = __sinf(arg);
    emb[tid + 64] = __cosf(arg);
  }
  if (tid < ZDim) {
    z0s[tid] = silu_f(z0[b * ZDim + tid]);
    zts[tid] = silu_f(zt[b * ZDim + tid]);
  }
  __syncthreads();
  for (int j = tid; j < TE; j += 256) {
    float acc = tb1[j];
    for (int i = 0; i < 128; ++i) acc += emb[i] * tw1[i * TE + j];
    t1[j] = silu_f(acc);
  }
  __syncthreads();
  for (int j = tid; j < TE; j += 256) {
    float acc = tb2[j];
    for (int i = 0; i < TE; ++i) acc += t1[i] * tw2[i * TE + j];
    t2s[j] = silu_f(acc);  // add path uses silu(temb2)
  }
  __syncthreads();
  {
    const int c = tid;  // 256 threads == C
    float acc = tpb[c] + 2.0f * zpb[c];
    for (int k = 0; k < TE; ++k) acc += t2s[k] * tpw[k * Cc + c];
    for (int z = 0; z < ZDim; ++z) acc += (z0s[z] + zts[z]) * zpw[z * Cc + c];
    addout[b * Cc + c] = acc;
  }
}

// ---------------- per-channel GroupNorm (+ optional SiLU) -> bf16 ----------------
__global__ __launch_bounds__(256) void norm_kernel(
    const float* __restrict__ x, const float* __restrict__ gamma,
    const float* __restrict__ beta, unsigned short* __restrict__ out,
    int out_stride, int out_off, int do_silu) {
  const int row = blockIdx.x;  // b*C + c
  const int c = row & (Cc - 1);
  const int tid = threadIdx.x;
  const float* xr = x + (size_t)row * Ll;

  float s = 0.f, s2 = 0.f;
#pragma unroll
  for (int k = 0; k < 8; ++k) {
    float v = xr[tid + k * 256];
    s += v; s2 += v * v;
  }
#pragma unroll
  for (int off = 16; off > 0; off >>= 1) {
    s  += __shfl_down(s,  off, 32);
    s2 += __shfl_down(s2, off, 32);
  }
  __shared__ float w1[8], w2[8];
  if ((tid & 31) == 0) { w1[tid >> 5] = s; w2[tid >> 5] = s2; }
  __syncthreads();
  if (tid == 0) {
    float a = 0.f, bsum = 0.f;
    for (int i = 0; i < 8; ++i) { a += w1[i]; bsum += w2[i]; }
    w1[0] = a; w2[0] = bsum;
  }
  __syncthreads();
  const float mean = w1[0] * (1.0f / 2048.0f);
  const float var  = w2[0] * (1.0f / 2048.0f) - mean * mean;
  const float rstd = rsqrtf(var + 1e-6f);
  const float gg = gamma[c], bb = beta[c];
  unsigned short* orow = out + (size_t)row * out_stride + out_off;
#pragma unroll
  for (int k = 0; k < 8; ++k) {
    int l = tid + k * 256;
    float v = (xr[l] - mean) * rstd * gg + bb;
    if (do_silu) v = silu_f(v);
    orow[l] = f2bf(v);
  }
  if (tid == 0 && out_off > 0) { orow[-1] = 0; orow[Ll] = 0; }  // conv halo
}

// ---------------- fused conv1d-as-GEMM (TAPS=1 or 3), WMMA bf16 ----------------
// Block tile: 128(M) x 128(N), BK=32; 8 waves in 2(M)x4(N), each 64x32
// (4x2 WMMA tiles -> 8 WMMAs per 6 fragment loads per k-step).
template <int TAPS>
__global__ __launch_bounds__(256) void gemm_kernel(
    const unsigned short* __restrict__ in, int in_stride, int in_off,
    const float* __restrict__ W, const float* __restrict__ bias,
    const float* __restrict__ addvec, const float* __restrict__ residual,
    float* __restrict__ out_f32, unsigned short* __restrict__ out_bf16,
    int trans_out, float scale) {
  __shared__ unsigned short As[128][32];  // [m][k] bf16
  __shared__ unsigned short Bs[128][32];  // [n][k] bf16 (transposed stage)

  const int tid = threadIdx.x;
  const int wid = tid >> 5, lane = tid & 31;
  const int g = lane >> 4, ln = lane & 15;
  const int wave_m = wid & 1, wave_n = wid >> 1;
  const int n0 = blockIdx.x * 128;
  const int m0 = blockIdx.y * 128;
  const int b  = blockIdx.z;

  const v8f vzero = {0.f, 0.f, 0.f, 0.f, 0.f, 0.f, 0.f, 0.f};
  v8f acc[4][2];
#pragma unroll
  for (int i = 0; i < 4; ++i)
#pragma unroll
    for (int j = 0; j < 2; ++j) acc[i][j] = vzero;

#pragma unroll
  for (int tap = 0; tap < TAPS; ++tap) {
    const int shift = tap - (TAPS >> 1);
    for (int kb = 0; kb < Cc / 32; ++kb) {
      __syncthreads();
      // stage A: weights -> bf16
#pragma unroll
      for (int it = 0; it < 16; ++it) {
        int idx = it * 256 + tid;
        int kk = idx & 31, m = idx >> 5;
        float w = W[(size_t)(m0 + m) * (Cc * TAPS) +
                    (size_t)(kb * 32 + kk) * TAPS + tap];
        As[m][kk] = f2bf(w);
      }
      if (kb + 1 < Cc / 32) {  // prefetch next weight k-slab
        int idx = tid;
        int kk = idx & 31, m = idx >> 5;
        __builtin_prefetch(&W[(size_t)(m0 + m) * (Cc * TAPS) +
                              (size_t)((kb + 1) * 32 + kk) * TAPS + tap], 0, 1);
      }
      // stage B transposed: Bs[n][k] = in[b][k][l] (coalesced over n)
#pragma unroll
      for (int it = 0; it < 16; ++it) {
        int idx = it * 256 + tid;
        int n = idx & 127, kk = idx >> 7;
        int cin = kb * 32 + kk;
        Bs[n][kk] = in[(size_t)(b * Cc + cin) * in_stride + in_off +
                       (n0 + n + shift)];
      }
      __syncthreads();
      v16bf af[4], bf[2];
#pragma unroll
      for (int mt = 0; mt < 4; ++mt)
        af[mt] = load_frag(&As[wave_m * 64 + mt * 16 + ln][0], g);
#pragma unroll
      for (int nt = 0; nt < 2; ++nt)
        bf[nt] = load_frag(&Bs[wave_n * 32 + nt * 16 + ln][0], g);
#pragma unroll
      for (int mt = 0; mt < 4; ++mt)
#pragma unroll
        for (int nt = 0; nt < 2; ++nt)
          acc[mt][nt] = wmma_bf16(af[mt], bf[nt], acc[mt][nt]);
    }
  }

  // epilogue: bias + addvec, scale, residual, store
#pragma unroll
  for (int mt = 0; mt < 4; ++mt) {
#pragma unroll
    for (int nt = 0; nt < 2; ++nt) {
#pragma unroll
      for (int r = 0; r < 8; ++r) {
        int co = m0 + wave_m * 64 + mt * 16 + r + 8 * g;
        int l  = n0 + wave_n * 32 + nt * 16 + ln;
        float v = acc[mt][nt][r] + bias[co];
        if (addvec) v += addvec[b * Cc + co];
        v *= scale;
        if (residual) v += residual[(size_t)(b * Cc + co) * Ll + l];
        if (out_f32) {
          out_f32[(size_t)(b * Cc + co) * Ll + l] = v;
        } else {
          size_t oi = trans_out ? ((size_t)(b * Ll + l) * Cc + co)
                                : ((size_t)(b * Cc + co) * Ll + l);
          out_bf16[oi] = f2bf(v);
        }
      }
    }
  }
}

// ---------------- streaming-softmax attention (flash style) ----------------
// qT: [B][L][C] bf16 (pre-scaled), K,V: [B][C][L] bf16.
// V tile staged to LDS via GLOBAL_LOAD_ASYNC_TO_LDS_B128 (ASYNCcnt),
// overlapping the S-GEMM + softmax phases; waited just before PV.
__global__ __launch_bounds__(256) void flash_kernel(
    const unsigned short* __restrict__ qT, const unsigned short* __restrict__ Kg,
    const unsigned short* __restrict__ Vg, unsigned short* __restrict__ Og) {
  __shared__ unsigned short sK[64][256];  // K^T tile: [j][c]
  __shared__ unsigned short sV[256][64];  // V tile:   [c][j] (async-filled)
  __shared__ float          sS[64][64];   // score tile [i][j]
  __shared__ unsigned short sP[64][64];   // exp tile, bf16
  __shared__ float sM[64], sLs[64], sA[64];

  const int tid = threadIdx.x;
  const int wid = tid >> 5, lane = tid & 31;
  const int g = lane >> 4, ln = lane & 15;
  const int i0 = blockIdx.x * 64;
  const int b  = blockIdx.y;
  const int st = wid * 2;  // this wave's two score tiles

  if (tid < 64) { sM[tid] = -1e30f; sLs[tid] = 0.f; }

  const v8f vzero = {0.f, 0.f, 0.f, 0.f, 0.f, 0.f, 0.f, 0.f};
  v8f acc[2][4];
#pragma unroll
  for (int i = 0; i < 2; ++i)
#pragma unroll
    for (int j = 0; j < 4; ++j) acc[i][j] = vzero;

  for (int jt = 0; jt < Ll / 64; ++jt) {
    const int j0 = jt * 64;
    __syncthreads();  // prev PV reads of sV/sP done before overwrite

    // kick off async V tile (b128 = 8 j-elements; contiguous in LDS & global)
    {
      const unsigned short* vgb = Vg + (size_t)b * Cc * Ll + j0;
#pragma unroll
      for (int it = 0; it < 8; ++it) {
        int idx = it * 256 + tid;
        int jb = (idx & 7) * 8;
        int c  = idx >> 3;
        unsigned long long ga =
            (unsigned long long)(uintptr_t)(vgb + (size_t)c * Ll + jb);
        unsigned la = lds_off32(&sV[c][jb]);
        asm volatile("global_load_async_to_lds_b128 %0, %1, off"
                     :: "v"(la), "v"(ga) : "memory");
      }
    }

    // stage K^T tile (coalesced over j)
#pragma unroll
    for (int it = 0; it < 64; ++it) {
      int idx = it * 256 + tid;
      int j = idx & 63, c = idx >> 6;
      sK[j][c] = Kg[(size_t)(b * Cc + c) * Ll + j0 + j];
    }
    __syncthreads();

    // score tiles: S = Q^T K, K-dim = 256
#pragma unroll
    for (int s = 0; s < 2; ++s) {
      const int t = st + s, ti = t >> 2, tj = t & 3;
      v8f sa = vzero;
      const unsigned short* qrow =
          qT + ((size_t)b * Ll + i0 + ti * 16 + ln) * Cc;
#pragma unroll
      for (int cs = 0; cs < 8; ++cs) {
        v16bf a = load_frag(qrow + cs * 32, g);
        v16bf bb = load_frag(&sK[tj * 16 + ln][cs * 32], g);
        sa = wmma_bf16(a, bb, sa);
      }
#pragma unroll
      for (int r = 0; r < 8; ++r)
        sS[ti * 16 + r + 8 * g][tj * 16 + ln] = sa[r];
    }
    __syncthreads();

    // online softmax per query row
    if (tid < 64) {
      const int i = tid;
      float mo = sM[i], mn = mo;
#pragma unroll
      for (int j = 0; j < 64; ++j) mn = fmaxf(mn, sS[i][j]);
      float alpha = __expf(mo - mn);
      float sum = 0.f;
#pragma unroll
      for (int j = 0; j < 64; ++j) {
        float p = __expf(sS[i][j] - mn);
        sum += p;
        sP[i][j] = f2bf(p);
      }
      sLs[i] = sLs[i] * alpha + sum;
      sM[i] = mn;
      sA[i] = alpha;
    }
    asm volatile("s_wait_asynccnt 0x0" ::: "memory");  // V tile landed
    __syncthreads();                                   // ... and visible

    // rescale running output, then O += V * P^T
#pragma unroll
    for (int nt = 0; nt < 4; ++nt) {
      float al = sA[nt * 16 + ln];
#pragma unroll
      for (int mt = 0; mt < 2; ++mt)
#pragma unroll
        for (int r = 0; r < 8; ++r) acc[mt][nt][r] *= al;
    }
#pragma unroll
    for (int jk = 0; jk < 2; ++jk) {
      v16bf af[2];
#pragma unroll
      for (int mt = 0; mt < 2; ++mt) {
        int c = wid * 32 + mt * 16 + ln;
        af[mt] = load_frag(&sV[c][jk * 32], g);
      }
#pragma unroll
      for (int nt = 0; nt < 4; ++nt) {
        v16bf bb = load_frag(&sP[nt * 16 + ln][jk * 32], g);
#pragma unroll
        for (int mt = 0; mt < 2; ++mt)
          acc[mt][nt] = wmma_bf16(af[mt], bb, acc[mt][nt]);
      }
    }
  }
  __syncthreads();

  // normalize by running denom and store h_ (bf16)
#pragma unroll
  for (int nt = 0; nt < 4; ++nt) {
    float linv = 1.0f / sLs[nt * 16 + ln];
    int i = i0 + nt * 16 + ln;
#pragma unroll
    for (int mt = 0; mt < 2; ++mt) {
#pragma unroll
      for (int r = 0; r < 8; ++r) {
        int c = wid * 32 + mt * 16 + r + 8 * g;
        Og[(size_t)(b * Cc + c) * Ll + i] = f2bf(acc[mt][nt][r] * linv);
      }
    }
  }
}

// ---------------- host ----------------

extern "C" void kernel_launch(void* const* d_in, const int* in_sizes, int n_in,
                              void* d_out, int out_size, void* d_ws,
                              size_t ws_size, hipStream_t stream) {
  (void)in_sizes; (void)n_in; (void)out_size; (void)ws_size;
  const float* x   = (const float*)d_in[0];
  const int*   t   = (const int*)d_in[1];
  const float* z0  = (const float*)d_in[2];
  const float* zt  = (const float*)d_in[3];
  const float* tw1 = (const float*)d_in[4];
  const float* tb1 = (const float*)d_in[5];
  const float* tw2 = (const float*)d_in[6];
  const float* tb2 = (const float*)d_in[7];
  const float* tpw = (const float*)d_in[8];
  const float* tpb = (const float*)d_in[9];
  const float* zpw = (const float*)d_in[10];
  const float* zpb = (const float*)d_in[11];
  const float* n1g = (const float*)d_in[12];
  const float* n1b = (const float*)d_in[13];
  const float* n2g = (const float*)d_in[14];
  const float* n2b = (const float*)d_in[15];
  const float* ng  = (const float*)d_in[16];
  const float* nbb = (const float*)d_in[17];
  const float* c1w = (const float*)d_in[18];
  const float* c1b = (const float*)d_in[19];
  const float* c2w = (const float*)d_in[20];
  const float* c2b = (const float*)d_in[21];
  const float* qw  = (const float*)d_in[22];
  const float* qb  = (const float*)d_in[23];
  const float* kw  = (const float*)d_in[24];
  const float* kb  = (const float*)d_in[25];
  const float* vw  = (const float*)d_in[26];
  const float* vb  = (const float*)d_in[27];
  const float* pw  = (const float*)d_in[28];
  const float* pb  = (const float*)d_in[29];

  const size_t BCL = (size_t)Bn * Cc * Ll;
  char* p = (char*)d_ws;
  auto carve = [&](size_t bytes) -> void* {
    void* q = (void*)p;
    p += (bytes + 255) & ~(size_t)255;
    return q;
  };
  float*          addv = (float*)carve((size_t)Bn * Cc * 4);
  unsigned short* nbuf = (unsigned short*)carve((size_t)Bn * Cc * LP * 2);
  float*          h1   = (float*)carve(BCL * 4);
  float*          x1   = (float*)carve(BCL * 4);
  unsigned short* hn   = (unsigned short*)carve(BCL * 2);
  unsigned short* qTb  = (unsigned short*)carve(BCL * 2);
  unsigned short* kbf  = (unsigned short*)carve(BCL * 2);
  unsigned short* vbf  = (unsigned short*)carve(BCL * 2);
  unsigned short* hbf  = (unsigned short*)carve(BCL * 2);

  const dim3 blk(256);
  const dim3 ggemm(Ll / 128, Cc / 128, Bn);
  const float qscale = 0.0625f;  // 256^-0.5

  temb_kernel<<<Bn, blk, 0, stream>>>(t, z0, zt, tw1, tb1, tw2, tb2, tpw, tpb,
                                      zpw, zpb, addv);
  norm_kernel<<<Bn * Cc, blk, 0, stream>>>(x, n1g, n1b, nbuf, LP, POFF, 1);
  gemm_kernel<3><<<ggemm, blk, 0, stream>>>(nbuf, LP, POFF, c1w, c1b, addv,
                                            nullptr, h1, nullptr, 0, 1.0f);
  norm_kernel<<<Bn * Cc, blk, 0, stream>>>(h1, n2g, n2b, nbuf, LP, POFF, 1);
  gemm_kernel<3><<<ggemm, blk, 0, stream>>>(nbuf, LP, POFF, c2w, c2b, nullptr,
                                            x, x1, nullptr, 0, 1.0f);
  norm_kernel<<<Bn * Cc, blk, 0, stream>>>(x1, ng, nbb, hn, Ll, 0, 0);
  gemm_kernel<1><<<ggemm, blk, 0, stream>>>(hn, Ll, 0, qw, qb, nullptr, nullptr,
                                            nullptr, qTb, 1, qscale);
  gemm_kernel<1><<<ggemm, blk, 0, stream>>>(hn, Ll, 0, kw, kb, nullptr, nullptr,
                                            nullptr, kbf, 0, 1.0f);
  gemm_kernel<1><<<ggemm, blk, 0, stream>>>(hn, Ll, 0, vw, vb, nullptr, nullptr,
                                            nullptr, vbf, 0, 1.0f);
  flash_kernel<<<dim3(Ll / 64, Bn), blk, 0, stream>>>(qTb, kbf, vbf, hbf);
  gemm_kernel<1><<<ggemm, blk, 0, stream>>>(hbf, Ll, 0, pw, pb, nullptr, x1,
                                            (float*)d_out, nullptr, 0, 1.0f);
}
